// GNN_35124242547080
// MI455X (gfx1250) — matrix operand order repaired and checked by reading, not compiled, not run on hardware
//
#include <hip/hip_runtime.h>
#include <hip/hip_bf16.h>

// ---------------- problem constants ----------------
#define NN      20000
#define EE      320000
#define GG      256
#define HH1     256
#define HH2     128
#define HEADS   4
#define RR      8
#define KBIG    2304          // R*G + G
#define OUTC    512           // HEADS*H2
#define EPS_BN  1e-5f

typedef __attribute__((ext_vector_type(16))) _Float16 v16h;
typedef __attribute__((ext_vector_type(8)))  _Float16 v8h;
typedef __attribute__((ext_vector_type(8)))  float    v8f;

// -------- monotonic float<->uint encoding for atomic max --------
__device__ __forceinline__ unsigned encf(float f) {
    unsigned u = __float_as_uint(f);
    return (u & 0x80000000u) ? ~u : (u | 0x80000000u);
}
__device__ __forceinline__ float decf(unsigned e) {
    unsigned u = (e & 0x80000000u) ? (e & 0x7FFFFFFFu) : ~e;
    return __uint_as_float(u);
}

// -------- gfx1250 async copy: global -> LDS, 16B per lane, ASYNCcnt --------
__device__ __forceinline__ void async_cp_b128(unsigned ldsoff,
                                              unsigned long long gaddr) {
    asm volatile("global_load_async_to_lds_b128 %0, %1, off"
                 :: "v"(ldsoff), "v"(gaddr)
                 : "memory");
}

// ---------------- convert x into tail columns of Ah (f16) ----------------
__global__ __launch_bounds__(256) void conv_x_kernel(
    const float* __restrict__ x, _Float16* __restrict__ Ah)
{
    int idx = blockIdx.x * 256 + threadIdx.x;        // N*G threads
    if (idx >= NN * GG) return;
    int n = idx >> 8, g = idx & 255;
    Ah[(size_t)n * KBIG + (RR * GG) + g] = (_Float16)x[idx];
}

// -------- scatter raw features into (dst, rel) bins (f32), count edges --------
__global__ __launch_bounds__(256) void scatter_feat_kernel(
    const float* __restrict__ x, const int* __restrict__ edge_index,
    const int* __restrict__ edge_type,
    float* __restrict__ S, float* __restrict__ cnt)
{
    int e = blockIdx.x;
    int t = threadIdx.x;                             // 0..255 == channel
    int src = edge_index[e];
    int dst = edge_index[EE + e];
    int rt  = edge_type[e];
    float val = x[(size_t)src * GG + t];
    atomicAdd(&S[(size_t)dst * (RR * GG) + (size_t)rt * GG + t], val);
    if (t == 0) atomicAdd(&cnt[dst * RR + rt], 1.0f);
}

// ------- divide bins by counts -> means, convert to f16 head of Ah -------
__global__ __launch_bounds__(256) void norm_conv_kernel(
    const float* __restrict__ S, const float* __restrict__ cnt,
    _Float16* __restrict__ Ah)
{
    int idx = blockIdx.x * 256 + threadIdx.x;        // N * (R*G) threads
    if (idx >= NN * RR * GG) return;
    int n = idx >> 11;                               // / 2048
    int c = idx & 2047;
    float k = cnt[n * RR + (c >> 8)];
    Ah[(size_t)n * KBIG + c] = (_Float16)(S[idx] * (1.0f / fmaxf(k, 1.0f)));
}

// ---- build transposed f16 RGCN weight: Bt[h][kk] = [W_rel;W_root](kk,h) ----
__global__ __launch_bounds__(256) void trans_rgcnB_kernel(
    const float* __restrict__ W_rel, const float* __restrict__ W_root,
    _Float16* __restrict__ Bt)
{
    int idx = blockIdx.x * 256 + threadIdx.x;        // HH1 * KBIG threads
    if (idx >= HH1 * KBIG) return;
    int h  = idx / KBIG;
    int kk = idx - h * KBIG;
    float v = (kk < RR * GG) ? W_rel[(size_t)kk * HH1 + h]
                             : W_root[(size_t)(kk - RR * GG) * HH1 + h];
    Bt[idx] = (_Float16)v;
}

// ---- transpose+convert projection weight: Bt[n][k] = W[k][n], K=256 ----
__global__ __launch_bounds__(256) void trans_w256_kernel(
    const float* __restrict__ W, _Float16* __restrict__ Bt, int N)
{
    int idx = blockIdx.x * 256 + threadIdx.x;        // N*256 threads
    if (idx >= N * 256) return;
    int n = idx >> 8, k = idx & 255;
    Bt[idx] = (_Float16)W[(size_t)k * N + n];
}

// ---------------- f16-WMMA GEMM:  C = A * B^T + bias ----------------
// A:  M x K row-major f16.   Bt: N x K row-major f16 (i.e. B transposed).
// Requires K % 32 == 0, N % 128 == 0 (true for all call sites).
// 256 threads (8 waves), 64x128 tile, K-step 32, double-buffered LDS
// filled by gfx1250 async global->LDS copies (ASYNCcnt pipelined).
#define TM 64
#define TN 128
#define TK 32
#define AST 40            // LDS row stride in halves (80B, 16B-aligned)
#define BST 40
template<int HOUT>
__global__ __launch_bounds__(256) void gemm_f16_wmma(
    const _Float16* __restrict__ A,
    const _Float16* __restrict__ Bt,
    const float* __restrict__ bias,
    float* __restrict__ C, _Float16* __restrict__ Ch, int ldc,
    int M, int N, int K)
{
    __shared__ __attribute__((aligned(16))) _Float16 As[2 * TM * AST];
    __shared__ __attribute__((aligned(16))) _Float16 Bp[2 * TN * BST];

    int tid  = threadIdx.x;
    int wave = tid >> 5;
    int lane = tid & 31;
    int bm   = blockIdx.x * TM;
    int bn   = blockIdx.y * TN;

    int wm  = (wave & 3) * 16;                 // subtile row in tile
    int wn0 = (wave >> 2) * 64;                // 4 subtile cols per wave

    int rfrag = lane & 15;
    int hsel  = lane >> 4;                     // 0 or 1

    // staging assignments (16B per async op per lane)
    int ar = tid >> 2;                         // A: 64 rows, 4 thr/row
    int ac = (tid & 3) * 8;                    // 8 halves each
    int br = tid >> 1;                         // B: 128 rows, 2 thr/row
    int bc = (tid & 1) * 16;                   // 16 halves each (2 ops)

    // clamp OOB A rows to M-1: duplicated data only feeds discarded C rows
    int amrow = bm + ar; if (amrow >= M) amrow = M - 1;
    unsigned long long aga =
        (unsigned long long)(size_t)(A + (size_t)amrow * K + ac);
    unsigned long long bga =
        (unsigned long long)(size_t)(Bt + (size_t)(bn + br) * K + bc);

    unsigned aoff[2], boff[2];
    #pragma unroll
    for (int b = 0; b < 2; ++b) {
        aoff[b] = (unsigned)(size_t)(As + b * (TM * AST) + ar * AST + ac);
        boff[b] = (unsigned)(size_t)(Bp + b * (TN * BST) + br * BST + bc);
    }

    v8f acc0 = {}, acc1 = {}, acc2 = {}, acc3 = {};

    int ksteps = K / TK;
    // prologue: tile 0 -> buffer 0
    {
        async_cp_b128(aoff[0], aga);
        async_cp_b128(boff[0], bga);
        async_cp_b128(boff[0] + 16, bga + 16);
    }

    for (int ks = 0; ks < ksteps; ++ks) {
        __syncthreads();        // everyone done reading buffer (ks+1)&1
        if (ks + 1 < ksteps) {  // issue tile ks+1 into the other buffer
            int b = (ks + 1) & 1;
            unsigned long long koff = (unsigned long long)(ks + 1) * (TK * 2);
            async_cp_b128(aoff[b], aga + koff);
            async_cp_b128(boff[b], bga + koff);
            async_cp_b128(boff[b] + 16, bga + koff + 16);
            asm volatile("s_wait_asynccnt 0x3" ::: "memory");
        } else {
            asm volatile("s_wait_asynccnt 0x0" ::: "memory");
        }
        __syncthreads();        // all waves' tile-ks copies landed

        const _Float16* Asb = As + (ks & 1) * (TM * AST);
        const _Float16* Bpb = Bp + (ks & 1) * (TN * BST);

        // ---- fragments: each is two ds_load_b128 ----
        v8h a0 = *(const v8h*)(Asb + (wm + rfrag) * AST + hsel * 16);
        v8h a1 = *(const v8h*)(Asb + (wm + rfrag) * AST + hsel * 16 + 8);
        v16h af = __builtin_shufflevector(a0, a1,
                    0,1,2,3,4,5,6,7,8,9,10,11,12,13,14,15);
        const _Float16* b0p = Bpb + (wn0 + rfrag) * BST + hsel * 16;
        v16h bf0 = __builtin_shufflevector(*(const v8h*)(b0p),
                                           *(const v8h*)(b0p + 8),
                    0,1,2,3,4,5,6,7,8,9,10,11,12,13,14,15);
        const _Float16* b1p = b0p + 16 * BST;
        v16h bf1 = __builtin_shufflevector(*(const v8h*)(b1p),
                                           *(const v8h*)(b1p + 8),
                    0,1,2,3,4,5,6,7,8,9,10,11,12,13,14,15);
        const _Float16* b2p = b0p + 32 * BST;
        v16h bf2 = __builtin_shufflevector(*(const v8h*)(b2p),
                                           *(const v8h*)(b2p + 8),
                    0,1,2,3,4,5,6,7,8,9,10,11,12,13,14,15);
        const _Float16* b3p = b0p + 48 * BST;
        v16h bf3 = __builtin_shufflevector(*(const v8h*)(b3p),
                                           *(const v8h*)(b3p + 8),
                    0,1,2,3,4,5,6,7,8,9,10,11,12,13,14,15);

        acc0 = __builtin_amdgcn_wmma_f32_16x16x32_f16(
            false, af, false, bf0, (short)0, acc0, false, false);
        acc1 = __builtin_amdgcn_wmma_f32_16x16x32_f16(
            false, af, false, bf1, (short)0, acc1, false, false);
        acc2 = __builtin_amdgcn_wmma_f32_16x16x32_f16(
            false, af, false, bf2, (short)0, acc2, false, false);
        acc3 = __builtin_amdgcn_wmma_f32_16x16x32_f16(
            false, af, false, bf3, (short)0, acc3, false, false);
    }

    // C/D layout: VGPR j -> M = j + (lane>=16 ? 8 : 0), N = lane&15
    int gm_base = bm + wm + (lane >> 4) * 8;
    int n_base  = bn + wn0 + (lane & 15);
    #pragma unroll
    for (int t = 0; t < 8; ++t) {
        int gm = gm_base + t;
        if (gm >= M) continue;
        #pragma unroll
        for (int j = 0; j < 4; ++j) {
            int n = n_base + j * 16;
            float val = (j == 0 ? acc0[t] : j == 1 ? acc1[t]
                       : j == 2 ? acc2[t] : acc3[t]) + bias[n];
            if (HOUT) Ch[(size_t)gm * ldc + n] = (_Float16)val;
            else      C [(size_t)gm * ldc + n] = val;
        }
    }
}

// ---------------- attention: per-edge/head dot + segment max ----------------
__global__ __launch_bounds__(256) void alpha_kernel(
    const float* __restrict__ q, const float* __restrict__ k,
    const int* __restrict__ edge_index,
    float* __restrict__ alpha, unsigned* __restrict__ amax)
{
    int wave = threadIdx.x >> 5;
    int lane = threadIdx.x & 31;
    int e = blockIdx.x * 2 + (wave >> 2);
    if (e >= EE) return;
    int h = wave & 3;
    int src = edge_index[e];
    int dst = edge_index[EE + e];
    const float* qp = q + (size_t)dst * OUTC + h * HH2;
    const float* kp = k + (size_t)src * OUTC + h * HH2;
    float s = 0.0f;
    #pragma unroll
    for (int c = 0; c < HH2; c += 32) s += qp[c + lane] * kp[c + lane];
    #pragma unroll
    for (int m = 16; m; m >>= 1) s += __shfl_xor(s, m, 32);
    if (lane == 0) {
        float a = s * 0.08838834764831845f;     // 1/sqrt(128)
        alpha[e * HEADS + h] = a;
        atomicMax(&amax[dst * HEADS + h], encf(a));
    }
}

// ------------- exp(alpha - max) and segment-sum denominators -------------
__global__ __launch_bounds__(256) void expsum_kernel(
    const int* __restrict__ edge_index, const unsigned* __restrict__ amax,
    float* __restrict__ alpha, float* __restrict__ denom)
{
    int idx = blockIdx.x * 256 + threadIdx.x;   // E*HEADS threads
    if (idx >= EE * HEADS) return;
    int e = idx >> 2, h = idx & 3;
    int dst = edge_index[EE + e];
    float m  = decf(amax[dst * HEADS + h]);
    float ea = __expf(alpha[idx] - m);
    alpha[idx] = ea;
    atomicAdd(&denom[dst * HEADS + h], ea);
}

// ---------------- weighted scatter of v[src] into out[dst] ----------------
__global__ __launch_bounds__(256) void scatter_attn_kernel(
    const float* __restrict__ v, const int* __restrict__ edge_index,
    const float* __restrict__ alpha, const float* __restrict__ denom,
    float* __restrict__ out)
{
    int e = blockIdx.x;
    int t = threadIdx.x;
    int src = edge_index[e];
    int dst = edge_index[EE + e];
    #pragma unroll
    for (int c = t; c < OUTC; c += 256) {
        int h = c >> 7;
        float w = alpha[e * HEADS + h] /
                  fmaxf(denom[dst * HEADS + h], 1e-16f);
        atomicAdd(&out[(size_t)dst * OUTC + c], w * v[(size_t)src * OUTC + c]);
    }
}

// ---------------- batch-norm column statistics ----------------
__global__ __launch_bounds__(256) void bn_stats_kernel(
    const float* __restrict__ out, float* __restrict__ mu, float* __restrict__ var)
{
    __shared__ float s1[256], s2[256];
    int col = blockIdx.x;                      // 0..511
    float a = 0.0f, b = 0.0f;
    for (int r = threadIdx.x; r < NN; r += 256) {
        float x = out[(size_t)r * OUTC + col];
        a += x; b += x * x;
    }
    s1[threadIdx.x] = a; s2[threadIdx.x] = b;
    __syncthreads();
    for (int st = 128; st; st >>= 1) {
        if (threadIdx.x < st) {
            s1[threadIdx.x] += s1[threadIdx.x + st];
            s2[threadIdx.x] += s2[threadIdx.x + st];
        }
        __syncthreads();
    }
    if (threadIdx.x == 0) {
        float m = s1[0] * (1.0f / NN);
        mu[col]  = m;
        var[col] = s2[0] * (1.0f / NN) - m * m;
    }
}

// ---------------- BN affine + LeakyReLU ----------------
__global__ __launch_bounds__(256) void bn_apply_kernel(
    const float* __restrict__ out, const float* __restrict__ mu,
    const float* __restrict__ var, const float* __restrict__ gamma,
    const float* __restrict__ beta, float* __restrict__ y)
{
    int idx = blockIdx.x * 256 + threadIdx.x;   // N*OUTC threads
    if (idx >= NN * OUTC) return;
    int c = idx & (OUTC - 1);
    float xn = (out[idx] - mu[c]) * rsqrtf(var[c] + EPS_BN);
    float o  = gamma[c] * xn + beta[c];
    y[idx] = (o > 0.0f) ? o : 0.01f * o;
}

// =====================================================================
extern "C" void kernel_launch(void* const* d_in, const int* in_sizes, int n_in,
                              void* d_out, int out_size, void* d_ws, size_t ws_size,
                              hipStream_t stream)
{
    const float* x       = (const float*)d_in[0];
    // d_in[1] node_type: unused by the reference computation
    const int*   eidx    = (const int*)d_in[2];
    const int*   etype   = (const int*)d_in[3];
    const float* W_rel   = (const float*)d_in[4];
    const float* W_root  = (const float*)d_in[5];
    const float* b_rgcn  = (const float*)d_in[6];
    const float* W_q     = (const float*)d_in[7];
    const float* b_q     = (const float*)d_in[8];
    const float* W_k     = (const float*)d_in[9];
    const float* b_k     = (const float*)d_in[10];
    const float* W_v     = (const float*)d_in[11];
    const float* b_v     = (const float*)d_in[12];
    const float* W_skip  = (const float*)d_in[13];
    const float* b_skip  = (const float*)d_in[14];
    const float* gamma   = (const float*)d_in[15];
    const float* beta    = (const float*)d_in[16];
    float* y = (float*)d_out;

    // ---------------- workspace layout ----------------
    char* ws = (char*)d_ws;
    size_t off = 0;
    auto take = [&](size_t bytes) -> char* {
        char* p = ws + off;
        off = (off + bytes + 255) & ~(size_t)255;
        return p;
    };
    float*     S     = (float*)take((size_t)NN * RR * GG * 4);   // 164 MB (f32 bins)
    _Float16*  Ah    = (_Float16*)take((size_t)NN * KBIG * 2);   // 92 MB
    float*     cnt   = (float*)take((size_t)NN * RR * 4);
    _Float16*  BtR   = (_Float16*)take((size_t)HH1 * KBIG * 2);  // RGCN B^T
    _Float16*  h1h   = (_Float16*)take((size_t)NN * HH1 * 2);
    _Float16*  BtQ   = (_Float16*)take((size_t)OUTC * HH1 * 2);
    _Float16*  BtK   = (_Float16*)take((size_t)OUTC * HH1 * 2);
    _Float16*  BtV   = (_Float16*)take((size_t)OUTC * HH1 * 2);
    _Float16*  BtS   = (_Float16*)take((size_t)OUTC * HH1 * 2);
    float*     alpha = (float*)take((size_t)EE * HEADS * 4);
    unsigned*  amax  = (unsigned*)take((size_t)NN * HEADS * 4);
    float*     denom = (float*)take((size_t)NN * HEADS * 4);
    float*     outb  = (float*)take((size_t)NN * OUTC * 4);
    float*     mu    = (float*)take(OUTC * 4);
    float*     var   = (float*)take(OUTC * 4);
    // q/k/v/skip (f32) reuse the S region once Ah is built (164 MB == S size)
    float* qb = S;
    float* kb = S + (size_t)NN * OUTC;
    float* vb = S + (size_t)2 * NN * OUTC;
    float* sb = S + (size_t)3 * NN * OUTC;

    // ---------------- phase 1: RGCN ----------------
    hipMemsetAsync(S, 0, (size_t)NN * RR * GG * 4, stream);
    hipMemsetAsync(cnt, 0, (size_t)NN * RR * 4, stream);

    conv_x_kernel<<<(NN * GG + 255) / 256, 256, 0, stream>>>(x, Ah);
    trans_rgcnB_kernel<<<(HH1 * KBIG + 255) / 256, 256, 0, stream>>>(W_rel, W_root, BtR);
    trans_w256_kernel<<<(OUTC * HH1 + 255) / 256, 256, 0, stream>>>(W_q, BtQ, OUTC);
    trans_w256_kernel<<<(OUTC * HH1 + 255) / 256, 256, 0, stream>>>(W_k, BtK, OUTC);
    trans_w256_kernel<<<(OUTC * HH1 + 255) / 256, 256, 0, stream>>>(W_v, BtV, OUTC);
    trans_w256_kernel<<<(OUTC * HH1 + 255) / 256, 256, 0, stream>>>(W_skip, BtS, OUTC);

    scatter_feat_kernel<<<EE, 256, 0, stream>>>(x, eidx, etype, S, cnt);
    norm_conv_kernel<<<(NN * RR * GG + 255) / 256, 256, 0, stream>>>(S, cnt, Ah);

    // h1 (f16 out) = Ah(20000x2304) * BtR^T + b_rgcn
    dim3 g1((NN + TM - 1) / TM, HH1 / TN);     // (313, 2)
    gemm_f16_wmma<1><<<g1, 256, 0, stream>>>(Ah, BtR, b_rgcn,
                                             nullptr, h1h, HH1, NN, HH1, KBIG);

    // ---------------- phase 2: projections (f32 out) ----------------
    dim3 g2((NN + TM - 1) / TM, OUTC / TN);    // (313, 4)
    gemm_f16_wmma<0><<<g2, 256, 0, stream>>>(h1h, BtQ, b_q, qb, nullptr, OUTC, NN, OUTC, HH1);
    gemm_f16_wmma<0><<<g2, 256, 0, stream>>>(h1h, BtK, b_k, kb, nullptr, OUTC, NN, OUTC, HH1);
    gemm_f16_wmma<0><<<g2, 256, 0, stream>>>(h1h, BtV, b_v, vb, nullptr, OUTC, NN, OUTC, HH1);
    gemm_f16_wmma<0><<<g2, 256, 0, stream>>>(h1h, BtS, b_skip, sb, nullptr, OUTC, NN, OUTC, HH1);

    // ---------------- phase 3: attention ----------------
    hipMemsetAsync(amax, 0, (size_t)NN * HEADS * 4, stream);   // 0 == encoded minimum
    hipMemsetAsync(denom, 0, (size_t)NN * HEADS * 4, stream);
    alpha_kernel<<<(EE + 1) / 2, 256, 0, stream>>>(qb, kb, eidx, alpha, amax);
    expsum_kernel<<<(EE * HEADS + 255) / 256, 256, 0, stream>>>(eidx, amax, alpha, denom);
    // out starts as the skip branch, attention accumulates on top
    hipMemcpyAsync(outb, sb, (size_t)NN * OUTC * 4, hipMemcpyDeviceToDevice, stream);
    scatter_attn_kernel<<<EE, 256, 0, stream>>>(vb, eidx, alpha, denom, outb);

    // ---------------- phase 4: batch-norm + leaky relu ----------------
    bn_stats_kernel<<<OUTC, 256, 0, stream>>>(outb, mu, var);
    bn_apply_kernel<<<(NN * OUTC + 255) / 256, 256, 0, stream>>>(
        outb, mu, var, gamma, beta, y);
}
